// CausalSelfAttention_82643760709865
// MI455X (gfx1250) — compile-verified, matrix operand order checked by reading, not run on hardware
//
#include <hip/hip_runtime.h>

typedef __attribute__((ext_vector_type(16))) _Float16 v16h;
typedef __attribute__((ext_vector_type(8)))  _Float16 v8h;
typedef __attribute__((ext_vector_type(8)))  float    v8f;
typedef __attribute__((ext_vector_type(4)))  float    v4f;

#define TSEQ  8192
#define CDIM  1024
#define NSEG  8
#define NHEAD 16
#define LSEG  1024
#define DH    64

// K index pattern for 16-bit A/B fragments (CDNA5 ISA):
// VGPR j<4 : K = 2j + 8h ; VGPR j>=4 : K = 16 + 2(j-4) + 8h   (h = lane>=16)
static __device__ __forceinline__ int kpat(int j, int h) {
    return (j < 4) ? (2 * j + 8 * h) : (16 + 2 * (j - 4) + 8 * h);
}

// Packed fragment load: a lane's 16 f16 are two contiguous 16 B runs.
static __device__ __forceinline__ v16h frag_ld(const _Float16* __restrict__ p0) {
    v8h lo = *(const v8h*)p0;
    v8h hi = *(const v8h*)(p0 + 16);
    return __builtin_shufflevector(lo, hi, 0, 1, 2, 3, 4, 5, 6, 7,
                                   8, 9, 10, 11, 12, 13, 14, 15);
}

static __device__ __forceinline__ v8f wmma_f16(v16h a, v16h b, v8f c) {
    return __builtin_amdgcn_wmma_f32_16x16x32_f16(false, a, false, b,
                                                  (short)0, c, false, false);
}

// Async copy of one contiguous 128 B V row (global) into a padded LDS row.
// Same INST_OFFSET applies to both LDS and global sides (ISA 08_async_tensor).
static __device__ __forceinline__ void stage_v_async(const _Float16* __restrict__ grow,
                                                     _Float16* lrow) {
    const unsigned vdst = (unsigned)(size_t)lrow;   // low 32 bits = LDS offset
    asm volatile(
        "global_load_async_to_lds_b128 %0, %1, off\n\t"
        "global_load_async_to_lds_b128 %0, %1, off offset:16\n\t"
        "global_load_async_to_lds_b128 %0, %1, off offset:32\n\t"
        "global_load_async_to_lds_b128 %0, %1, off offset:48\n\t"
        "global_load_async_to_lds_b128 %0, %1, off offset:64\n\t"
        "global_load_async_to_lds_b128 %0, %1, off offset:80\n\t"
        "global_load_async_to_lds_b128 %0, %1, off offset:96\n\t"
        "global_load_async_to_lds_b128 %0, %1, off offset:112"
        :: "v"(vdst), "v"(grow) : "memory");
}

// ---------------------------------------------------------------------------
// One-time f32 -> f16 conversion
// ---------------------------------------------------------------------------
__global__ __launch_bounds__(256)
void f32_to_f16(const float* __restrict__ src, _Float16* __restrict__ dst, int n) {
    const int idx = (blockIdx.x * 256 + threadIdx.x) * 8;
    if (idx + 8 > n) return;
    v4f a = *(const v4f*)(src + idx);
    v4f b = *(const v4f*)(src + idx + 4);
    v8h o;
#pragma unroll
    for (int i = 0; i < 4; ++i) { o[i] = (_Float16)a[i]; o[i + 4] = (_Float16)b[i]; }
    *(v8h*)(dst + idx) = o;
}

// ---------------------------------------------------------------------------
// GEMM: out[M,N] = A[M,K](f16) @ W[N,K]^T(f16) + bias[N](f32)
// One wave computes a 64x64 tile: 4 A-frags x 4 B-frags -> 16 WMMAs / k-step.
// ---------------------------------------------------------------------------
template <bool OUT_F16>
__global__ __launch_bounds__(32)
void gemm64_f16(const _Float16* __restrict__ A, const _Float16* __restrict__ W,
                const float* __restrict__ bias, void* __restrict__ outp,
                int M, int N, int K) {
    const int n0 = blockIdx.x * 64;
    const int m0 = blockIdx.y * 64;
    const int lane = threadIdx.x & 31;
    const int nl = lane & 15, h = lane >> 4;

    v8f zero = {};
    v8f acc[4][4];
#pragma unroll
    for (int mi = 0; mi < 4; ++mi)
#pragma unroll
        for (int ni = 0; ni < 4; ++ni) acc[mi][ni] = zero;

    const _Float16* ap = A + (long)(m0 + nl) * K + 8 * h;
    const _Float16* bp = W + (long)(n0 + nl) * K + 8 * h;
    const long rstep = (long)16 * K;

    for (int k0 = 0; k0 < K; k0 += 32) {
        v16h af[4], bf[4];
#pragma unroll
        for (int t = 0; t < 4; ++t) af[t] = frag_ld(ap + t * rstep + k0);
#pragma unroll
        for (int t = 0; t < 4; ++t) bf[t] = frag_ld(bp + t * rstep + k0);
        // prefetch both streams ~16 k-steps ahead (global_prefetch_b8)
        __builtin_prefetch(ap + k0 + 512, 0, 0);
        __builtin_prefetch(bp + k0 + 512, 0, 0);
#pragma unroll
        for (int mi = 0; mi < 4; ++mi)
#pragma unroll
            for (int ni = 0; ni < 4; ++ni)
                acc[mi][ni] = wmma_f16(af[mi], bf[ni], acc[mi][ni]);
    }

#pragma unroll
    for (int ni = 0; ni < 4; ++ni) {
        const int col = n0 + ni * 16 + nl;
        const float bv = bias[col];
#pragma unroll
        for (int mi = 0; mi < 4; ++mi) {
#pragma unroll
            for (int i = 0; i < 8; ++i) {
                const long row = m0 + mi * 16 + i + 8 * h;
                const float val = acc[mi][ni][i] + bv;
                if (OUT_F16) ((_Float16*)outp)[row * N + col] = (_Float16)val;
                else         ((float*)outp)[row * N + col] = val;
            }
        }
    }
}

// ---------------------------------------------------------------------------
// Attention: one 32-key block step (MASKED only for the diagonal block).
// V block for this step was staged asynchronously into sVcur by the caller.
// ---------------------------------------------------------------------------
template <bool MASKED>
static __device__ __forceinline__ void attn_block(
    int kb, int q0, int ld,
    const _Float16* __restrict__ kbase,
    const v16h& qa0, const v16h& qa1,
    v8f (&o)[4], float (&mrow)[8], float (&lrow)[8],
    _Float16* sP, const _Float16* sVcur, int nl, int h) {
    v8f zero = {};

    // ---- S = Q K^T (Q pre-scaled by Dh^-0.5), optional causal mask --------
    v8f s[2];
#pragma unroll
    for (int sub = 0; sub < 2; ++sub) {
        const _Float16* kp = kbase + (long)(kb + sub * 16 + nl) * ld + 8 * h;
        v16h b0 = frag_ld(kp);
        v16h b1 = frag_ld(kp + 32);
        v8f t = zero;
        t = wmma_f16(qa0, b0, t);
        t = wmma_f16(qa1, b1, t);
        s[sub] = t;
    }
    if (MASKED) {
#pragma unroll
        for (int i = 0; i < 8; ++i) {
            const int m = q0 + i + 8 * h;
#pragma unroll
            for (int sub = 0; sub < 2; ++sub) {
                const int n = kb + sub * 16 + nl;
                if (n > m) s[sub][i] = -__builtin_inff();
            }
        }
    }

    // ---- online softmax (xor reductions stay within 16-lane halves) -------
    float alpha[8];
#pragma unroll
    for (int i = 0; i < 8; ++i) {
        float t = fmaxf(s[0][i], s[1][i]);
#pragma unroll
        for (int msk = 1; msk < 16; msk <<= 1)
            t = fmaxf(t, __shfl_xor(t, msk, 32));
        const float mnew = fmaxf(mrow[i], t);
        alpha[i] = __expf(mrow[i] - mnew);
        mrow[i] = mnew;
        const float p0 = __expf(s[0][i] - mnew);
        const float p1 = __expf(s[1][i] - mnew);
        s[0][i] = p0; s[1][i] = p1;
        float rs = p0 + p1;
#pragma unroll
        for (int msk = 1; msk < 16; msk <<= 1)
            rs += __shfl_xor(rs, msk, 32);
        lrow[i] = lrow[i] * alpha[i] + rs;
    }
#pragma unroll
    for (int d = 0; d < 4; ++d)
#pragma unroll
        for (int i = 0; i < 8; ++i) o[d][i] *= alpha[i];

    // ---- P (C-layout) -> LDS, reload as A-fragment -------------------------
#pragma unroll
    for (int i = 0; i < 8; ++i)
#pragma unroll
        for (int sub = 0; sub < 2; ++sub)
            sP[(i + 8 * h) * 32 + sub * 16 + nl] = (_Float16)s[sub][i];
    asm volatile("s_wait_dscnt 0" ::: "memory");   // same-wave DS RAW
    v16h pa;
#pragma unroll
    for (int j = 0; j < 8; ++j) {
        const int k = kpat(j, h);
        pa[2 * j]     = sP[nl * 32 + k];
        pa[2 * j + 1] = sP[nl * 32 + k + 1];
    }

    // ---- ensure async V copy landed, then O += P @ V -----------------------
    asm volatile("s_wait_asynccnt 0" ::: "memory");
#pragma unroll
    for (int d = 0; d < 4; ++d) {
        v16h bv;
#pragma unroll
        for (int j = 0; j < 8; ++j) {
            const int k = kpat(j, h);
            bv[2 * j]     = sVcur[k * 72 + d * 16 + nl];
            bv[2 * j + 1] = sVcur[(k + 1) * 72 + d * 16 + nl];
        }
        o[d] = wmma_f16(pa, bv, o[d]);
    }
    asm volatile("" ::: "memory");
}

// ---------------------------------------------------------------------------
// Flash-style causal attention over f16 qkv [T, 3C].
// One wave per (seg, head, 16-row q tile); 32-key blocks, online softmax.
// V blocks double-buffered in LDS via GLOBAL_LOAD_ASYNC_TO_LDS_B128.
// ---------------------------------------------------------------------------
__global__ __launch_bounds__(32)
void attn_flash(const _Float16* __restrict__ qkv, _Float16* __restrict__ y) {
    __shared__ _Float16 sP[16 * 32];          // P transpose staging (1 KB)
    __shared__ _Float16 sV[2][32 * 72];       // V double buffer (9 KB)

    const int q0   = blockIdx.x * 16;
    const int head = blockIdx.y;
    const int seg  = blockIdx.z;
    const int lane = threadIdx.x & 31;
    const int nl = lane & 15, h = lane >> 4;

    const int ld = 3 * CDIM;
    const _Float16* qbase = qkv + (long)seg * LSEG * ld + head * DH;
    const _Float16* kbase = qbase + CDIM;
    const _Float16* vbase = qbase + 2 * CDIM;

    // Q fragments, pre-scaled by Dh^-0.5 = 0.125 (packed f16 muls, once)
    v16h qa0 = frag_ld(qbase + (long)(q0 + nl) * ld + 0 + 8 * h);
    v16h qa1 = frag_ld(qbase + (long)(q0 + nl) * ld + 32 + 8 * h);
    qa0 = qa0 * (_Float16)0.125f;
    qa1 = qa1 * (_Float16)0.125f;

    v8f zero = {};
    v8f o[4];
#pragma unroll
    for (int d = 0; d < 4; ++d) o[d] = zero;
    float mrow[8], lrow[8];
#pragma unroll
    for (int i = 0; i < 8; ++i) { mrow[i] = -__builtin_inff(); lrow[i] = 0.0f; }

    const int kbm = q0 & ~31;                 // diagonal (masked) block start
    int buf = 0;
    // kick off async V copy for the first block (1 padded row per lane)
    stage_v_async(vbase + (long)(0 + lane) * ld, &sV[0][lane * 72]);

    for (int kb = 0; kb < kbm; kb += 32) {    // full, unmasked blocks
        // issue next block's V copy; it overlaps S/softmax of this block
        stage_v_async(vbase + (long)(kb + 32 + lane) * ld, &sV[buf ^ 1][lane * 72]);
        attn_block<false>(kb, q0, ld, kbase, qa0, qa1,
                          o, mrow, lrow, sP, sV[buf], nl, h);
        buf ^= 1;
    }
    attn_block<true>(kbm, q0, ld, kbase, qa0, qa1,
                     o, mrow, lrow, sP, sV[buf], nl, h);

    // ---- epilogue: O / l -> y[T, C] (f16) ---------------------------------
#pragma unroll
    for (int d = 0; d < 4; ++d) {
#pragma unroll
        for (int i = 0; i < 8; ++i) {
            const long row = (long)seg * LSEG + q0 + i + 8 * h;
            y[row * CDIM + head * DH + d * 16 + nl] = (_Float16)(o[d][i] / lrow[i]);
        }
    }
}

// ---------------------------------------------------------------------------
extern "C" void kernel_launch(void* const* d_in, const int* in_sizes, int n_in,
                              void* d_out, int out_size, void* d_ws, size_t ws_size,
                              hipStream_t stream) {
    (void)in_sizes; (void)n_in; (void)out_size; (void)ws_size;
    const float* x      = (const float*)d_in[0];
    const float* W_attn = (const float*)d_in[1];
    const float* b_attn = (const float*)d_in[2];
    const float* W_proj = (const float*)d_in[3];
    const float* b_proj = (const float*)d_in[4];
    float* out = (float*)d_out;

    _Float16* xh   = (_Float16*)d_ws;                        // [T, C]
    _Float16* Wah  = xh   + (size_t)TSEQ * CDIM;             // [3C, C]
    _Float16* Wph  = Wah  + (size_t)3 * CDIM * CDIM;         // [C, C]
    _Float16* qkvh = Wph  + (size_t)CDIM * CDIM;             // [T, 3C]
    _Float16* yh   = qkvh + (size_t)TSEQ * 3 * CDIM;         // [T, C]

    const int nx = TSEQ * CDIM, na = 3 * CDIM * CDIM, np = CDIM * CDIM;
    f32_to_f16<<<nx / (8 * 256), 256, 0, stream>>>(x, xh, nx);
    f32_to_f16<<<na / (8 * 256), 256, 0, stream>>>(W_attn, Wah, na);
    f32_to_f16<<<np / (8 * 256), 256, 0, stream>>>(W_proj, Wph, np);

    // 1) fused QKV projection -> f16 qkv
    dim3 g1((3 * CDIM) / 64, TSEQ / 64);
    gemm64_f16<true><<<g1, 32, 0, stream>>>(xh, Wah, b_attn, qkvh,
                                            TSEQ, 3 * CDIM, CDIM);
    // 2) per-segment causal attention -> f16 y
    dim3 g2(LSEG / 16, NHEAD, NSEG);
    attn_flash<<<g2, 32, 0, stream>>>(qkvh, yh);
    // 3) output projection -> f32 out
    dim3 g3(CDIM / 64, TSEQ / 64);
    gemm64_f16<false><<<g3, 32, 0, stream>>>(yh, Wph, b_proj, out,
                                             TSEQ, CDIM, CDIM);
}
